// RDFNSMultiHeadAttention_85117661872612
// MI455X (gfx1250) — matrix operand order repaired and checked by reading, not compiled, not run on hardware
//
#include <hip/hip_runtime.h>
#include <hip/hip_bf16.h>

#define DEVINL __device__ __forceinline__

typedef __bf16 bf16_t;
typedef __attribute__((ext_vector_type(16))) __bf16 v16bf;
typedef __attribute__((ext_vector_type(8)))  float  v8f;
typedef unsigned int u32x4 __attribute__((ext_vector_type(4)));
typedef int          i32x8 __attribute__((ext_vector_type(8)));
typedef int          i32x4 __attribute__((ext_vector_type(4)));

// ---------------------------------------------------------------------------
// WMMA helpers (CDNA5 wave32, 16x16x32 bf16 -> f32)
// ---------------------------------------------------------------------------
DEVINL v8f wmma_bf16(v16bf a, v16bf b, v8f c) {
  return __builtin_amdgcn_wmma_f32_16x16x32_bf16(
      false, a, false, b, (short)0, c, false, false);
}

DEVINL v16bf frag_a_ld(const bf16_t* t, int ld) {
  const int lane = threadIdx.x & 31;
  const int row = lane & 15;
  const int hi  = lane >> 4;
  v16bf a;
#pragma unroll
  for (int j = 0; j < 16; ++j) {
    const int k = (hi << 4) + j;
    a[j] = t[row * ld + k];
  }
  return a;
}

DEVINL v16bf frag_b_ld(const bf16_t* t, int ld) {
  const int lane = threadIdx.x & 31;
  const int col = lane & 15;
  const int hi  = lane >> 4;
  v16bf b;
#pragma unroll
  for (int j = 0; j < 16; ++j) b[j] = t[col * ld + (hi << 4) + j];
  return b;
}

// ---------------------------------------------------------------------------
// Tensor Data Mover: 2D tile load (bf16 elements) with LDS padding.
// Descriptor layout per CDNA5 ISA.
// This toolchain exposes the 6-arg builtin:
//   (u32x4 g0, i32x8 g1, i32x4 g2, i32x4 g3, i32x8 extra, i32 cpol)
// ---------------------------------------------------------------------------
DEVINL unsigned lds_off(const void* p) {
  // generic pointers to LDS carry the LDS offset in the low 32 bits
  return (unsigned)(unsigned long long)p;
}

DEVINL void tdm_load_2d_bf16(unsigned lds_addr, const void* gptr,
                             unsigned tile_k, unsigned tile_rows,
                             unsigned tensor_d0, unsigned tensor_d1,
                             unsigned long long stride0_elems,
                             unsigned pad_en, unsigned pad_interval_code,
                             unsigned pad_amount_code) {
  const unsigned long long ga = (unsigned long long)gptr;
  u32x4 g0;
  g0[0] = 1u;                                        // count=1 (valid), user mode
  g0[1] = lds_addr;                                  // lds_addr
  g0[2] = (unsigned)(ga & 0xFFFFFFFFu);              // global_addr[31:0]
  g0[3] = (unsigned)((ga >> 32) & 0x1FFFFFFu)        // global_addr[56:32]
        | (2u << 30);                                // type = 2 (image)
  i32x8 g1;
  g1[0] = (int)((1u << 16)                           // data_size = 1 -> 2 bytes
        | (pad_en << 20)
        | (pad_interval_code << 22)
        | (pad_amount_code << 25));
  g1[1] = (int)((tensor_d0 & 0xFFFFu) << 16);        // tensor_dim0[15:0]
  g1[2] = (int)((tensor_d0 >> 16) | ((tensor_d1 & 0xFFFFu) << 16));
  g1[3] = (int)((tensor_d1 >> 16) | (tile_k << 16)); // tile_dim0
  g1[4] = (int)(tile_rows & 0xFFFFu);                // tile_dim1 (tile_dim2=0)
  g1[5] = (int)(unsigned)(stride0_elems & 0xFFFFFFFFu);
  g1[6] = (int)(unsigned)((stride0_elems >> 32) & 0xFFFFu); // stride1 = 0
  g1[7] = 0;
  i32x4 g2 = {1, 0, 0, 0};                           // tensor_dim2=1, unused
  i32x4 g3 = {0, 0, 0, 0};
  i32x8 g4 = {0, 0, 0, 0, 0, 0, 0, 0};
  __builtin_amdgcn_tensor_load_to_lds(g0, g1, g2, g3, g4, 0);
}

// ---------------------------------------------------------------------------
// Problem constants
// ---------------------------------------------------------------------------
#define BATCH 2
#define NTOK  2048
#define EMB   1024
#define HEADS 16
#define HD    64
#define BH    (BATCH * HEADS)
#define MROWS (BATCH * NTOK)   // 4096

// ---------------------------------------------------------------------------
// f32 -> bf16 conversion
// ---------------------------------------------------------------------------
__global__ void cvt_bf16_kernel(const float* __restrict__ in,
                                bf16_t* __restrict__ out, int n) {
  int i = blockIdx.x * blockDim.x + threadIdx.x;
  if (i < n) out[i] = (bf16_t)in[i];
}

// ---------------------------------------------------------------------------
// Row sum of squares over head dim (64) for q2/k2
// ---------------------------------------------------------------------------
__global__ void rowsumsq_kernel(const bf16_t* __restrict__ in,
                                float* __restrict__ out, int rows) {
  int r = blockIdx.x * blockDim.x + threadIdx.x;
  if (r >= rows) return;
  const bf16_t* p = in + (size_t)r * HD;
  float s = 0.f;
#pragma unroll
  for (int j = 0; j < HD; ++j) { float v = (float)p[j]; s += v * v; }
  out[r] = s;
}

// ---------------------------------------------------------------------------
// Tiled WMMA GEMM: C[M x 1024] = A[M x 1024] * W^T + bias
// 128x64 block tile, 8 waves each owning 32x32 (4 WMMA per K-chunk).
// Tile staging via TDM (tensor_load_to_lds) with double buffering: the next
// chunk's DMA overlaps the current chunk's WMMAs; wave 0 issues and gates with
// s_wait_tensorcnt.
// mode 0: store bf16 head layout (B,H,N,D)            (Q, K)
// mode 1: store bf16 transposed head layout (B,H,D,N) (V)
// mode 3: store f32 row-major [M x 1024]              (final projection)
// ---------------------------------------------------------------------------
#define LDT 40   // 32 elems + 8 pad -> TDM pad: 16 DW interval (code 3), 4 DW (code 3)
__global__ __launch_bounds__(256) void gemm128_kernel(
    const bf16_t* __restrict__ A, const bf16_t* __restrict__ W,
    const float* __restrict__ bias, void* __restrict__ out, int mode) {
  __shared__ bf16_t As[2][128 * LDT];
  __shared__ bf16_t Bs[2][64 * LDT];
  const int m0 = blockIdx.x * 128;
  const int n0 = blockIdx.y * 64;
  const int t = threadIdx.x;
  const int w = t >> 5, lane = t & 31, col = lane & 15, hi = lane >> 4;
  const int mi0 = (w & 3) << 1;   // 2 row subtiles (32 rows)
  const int ni0 = (w >> 2) << 1;  // 2 col subtiles (32 cols)
  v8f acc[2][2] = {{{}, {}}, {{}, {}}};

  const bf16_t* Abase = A + (size_t)m0 * EMB;
  const bf16_t* Wbase = W + (size_t)n0 * EMB;

  if (t < 32) {  // wave 0 issues DMA for the first K-chunk
    tdm_load_2d_bf16(lds_off(&As[0][0]), Abase, 32, 128, EMB, 65536,
                     (unsigned long long)EMB, 1u, 3u, 3u);
    tdm_load_2d_bf16(lds_off(&Bs[0][0]), Wbase, 32, 64, EMB, 65536,
                     (unsigned long long)EMB, 1u, 3u, 3u);
  }

  int buf = 0;
  for (int kt = 0; kt < EMB; kt += 32) {
    if (t < 32) __builtin_amdgcn_s_wait_tensorcnt(0);
    __syncthreads();
    if (kt + 32 < EMB && t < 32) {  // prefetch next chunk into other buffer
      tdm_load_2d_bf16(lds_off(&As[buf ^ 1][0]), Abase + kt + 32, 32, 128,
                       EMB, 65536, (unsigned long long)EMB, 1u, 3u, 3u);
      tdm_load_2d_bf16(lds_off(&Bs[buf ^ 1][0]), Wbase + kt + 32, 32, 64,
                       EMB, 65536, (unsigned long long)EMB, 1u, 3u, 3u);
    }
    v16bf a0 = frag_a_ld(&As[buf][0] + (mi0 + 0) * 16 * LDT, LDT);
    v16bf a1 = frag_a_ld(&As[buf][0] + (mi0 + 1) * 16 * LDT, LDT);
    v16bf b0 = frag_b_ld(&Bs[buf][0] + (ni0 + 0) * 16 * LDT, LDT);
    v16bf b1 = frag_b_ld(&Bs[buf][0] + (ni0 + 1) * 16 * LDT, LDT);
    acc[0][0] = wmma_bf16(a0, b0, acc[0][0]);
    acc[0][1] = wmma_bf16(a0, b1, acc[0][1]);
    acc[1][0] = wmma_bf16(a1, b0, acc[1][0]);
    acc[1][1] = wmma_bf16(a1, b1, acc[1][1]);
    __syncthreads();
    buf ^= 1;
  }

#pragma unroll
  for (int im = 0; im < 2; ++im) {
#pragma unroll
    for (int in = 0; in < 2; ++in) {
      const int nglob = n0 + (ni0 + in) * 16 + col;
      const float bv = bias[nglob];
#pragma unroll
      for (int r = 0; r < 8; ++r) {
        const int mglob = m0 + (mi0 + im) * 16 + r + (hi << 3);
        const float v = acc[im][in][r] + bv;
        if (mode == 0) {
          const int b = mglob >> 11, n = mglob & 2047;
          const int h = nglob >> 6, d = nglob & 63;
          ((bf16_t*)out)[(((size_t)(b * HEADS + h) * NTOK + n) << 6) + d] =
              (bf16_t)v;
        } else if (mode == 1) {
          const int b = mglob >> 11, n = mglob & 2047;
          const int h = nglob >> 6, d = nglob & 63;
          ((bf16_t*)out)[((size_t)(b * HEADS + h) * HD + d) * NTOK + n] =
              (bf16_t)v;
        } else {
          ((float*)out)[(size_t)mglob * EMB + nglob] = v;
        }
      }
    }
  }
}

// ---------------------------------------------------------------------------
// Column-sum pass: ncol[bh][m] = sum_n exp(-max(d2,0)/64)
// ---------------------------------------------------------------------------
__global__ __launch_bounds__(256) void colsum_kernel(
    const bf16_t* __restrict__ qh, const bf16_t* __restrict__ kh,
    const float* __restrict__ q2, const float* __restrict__ k2,
    float* __restrict__ ncol) {
  const int bh = blockIdx.x;
  const int m0 = blockIdx.y * 64;
  const int t = threadIdx.x, w = t >> 5, lane = t & 31;
  const int col = lane & 15, hi = lane >> 4;
  __shared__ float csum[64];
  if (t < 64) csum[t] = 0.f;
  __syncthreads();

  const bf16_t* qb = qh + (size_t)bh * NTOK * HD;
  const bf16_t* kb = kh + (size_t)bh * NTOK * HD;

  v16bf bfr[4][2];
  float k2m[4];
#pragma unroll
  for (int mi = 0; mi < 4; ++mi) {
    const bf16_t* kt = kb + (size_t)(m0 + mi * 16) * HD;
    bfr[mi][0] = frag_b_ld(kt, HD);
    bfr[mi][1] = frag_b_ld(kt + 32, HD);
    k2m[mi] = k2[bh * NTOK + m0 + mi * 16 + col];
  }

  for (int nc = 0; nc < NTOK; nc += 128) {
    const int nw = nc + w * 16;
    v16bf a0 = frag_a_ld(qb + (size_t)nw * HD, HD);
    v16bf a1 = frag_a_ld(qb + (size_t)nw * HD + 32, HD);
    float q2r[8];
#pragma unroll
    for (int r = 0; r < 8; ++r) q2r[r] = q2[bh * NTOK + nw + r + (hi << 3)];
#pragma unroll
    for (int mi = 0; mi < 4; ++mi) {
      v8f c = {};
      c = wmma_bf16(a0, bfr[mi][0], c);
      c = wmma_bf16(a1, bfr[mi][1], c);
      float s = 0.f;
#pragma unroll
      for (int r = 0; r < 8; ++r) {
        float d2 = fmaxf(q2r[r] + k2m[mi] - 2.f * c[r], 0.f);
        s += __expf(-d2 * 0.015625f);
      }
      atomicAdd(&csum[mi * 16 + col], s);
    }
  }
  __syncthreads();
  if (t < 64) ncol[bh * NTOK + m0 + t] = csum[t];
}

// ---------------------------------------------------------------------------
// Fused pass: for 64 rows n, sweep m: p = exp(-d2/64)*rsqrt(ncol[m]),
// Z[n] += p, acc += P @ V. Epilogue: out = acc / Z -> ob (bf16 row-major).
// ---------------------------------------------------------------------------
#define LDP 72
__global__ __launch_bounds__(256) void fusedpv_kernel(
    const bf16_t* __restrict__ qh, const bf16_t* __restrict__ kh,
    const bf16_t* __restrict__ vT, const float* __restrict__ q2,
    const float* __restrict__ k2, const float* __restrict__ ncol,
    bf16_t* __restrict__ ob) {
  const int bh = blockIdx.x;
  const int n0 = blockIdx.y * 64;
  const int t = threadIdx.x, w = t >> 5, lane = t & 31;
  const int col = lane & 15, hi = lane >> 4;
  const int ni = w & 3;
  const int mi0 = (w >> 2) << 1;
  const int di0 = (w >> 2) << 1;

  __shared__ bf16_t P[64 * LDP];
  __shared__ float Zrow[64];
  __shared__ float rsm[64];
  if (t < 64) Zrow[t] = 0.f;

  const bf16_t* qb = qh + (size_t)bh * NTOK * HD;
  const bf16_t* kb = kh + (size_t)bh * NTOK * HD;
  const bf16_t* vb = vT + (size_t)bh * HD * NTOK;

  v16bf qa0 = frag_a_ld(qb + (size_t)(n0 + ni * 16) * HD, HD);
  v16bf qa1 = frag_a_ld(qb + (size_t)(n0 + ni * 16) * HD + 32, HD);
  float q2r[8];
#pragma unroll
  for (int r = 0; r < 8; ++r)
    q2r[r] = q2[bh * NTOK + n0 + ni * 16 + r + (hi << 3)];

  v8f acc0 = {}, acc1 = {};
  __syncthreads();

  for (int m0 = 0; m0 < NTOK; m0 += 64) {
    if (t < 64) rsm[t] = rsqrtf(fmaxf(ncol[bh * NTOK + m0 + t], 1e-30f));
    __syncthreads();

#pragma unroll
    for (int e = 0; e < 2; ++e) {
      const int mi = mi0 + e;
      const bf16_t* kt = kb + (size_t)(m0 + mi * 16) * HD;
      v16bf b0 = frag_b_ld(kt, HD);
      v16bf b1 = frag_b_ld(kt + 32, HD);
      v8f c = {};
      c = wmma_bf16(qa0, b0, c);
      c = wmma_bf16(qa1, b1, c);
      const float k2m = k2[bh * NTOK + m0 + mi * 16 + col];
      const float rs = rsm[mi * 16 + col];
#pragma unroll
      for (int r = 0; r < 8; ++r) {
        float d2 = fmaxf(q2r[r] + k2m - 2.f * c[r], 0.f);
        float p = __expf(-d2 * 0.015625f) * rs;
        const int rl = ni * 16 + r + (hi << 3);
        atomicAdd(&Zrow[rl], p);
        P[rl * LDP + mi * 16 + col] = (bf16_t)p;
      }
    }
    __syncthreads();

    v16bf pa0 = frag_a_ld(P + ni * 16 * LDP, LDP);
    v16bf pa1 = frag_a_ld(P + ni * 16 * LDP + 32, LDP);
    {
      const bf16_t* vt0 = vb + (size_t)(di0 * 16) * NTOK + m0;
      const bf16_t* vt1 = vb + (size_t)((di0 + 1) * 16) * NTOK + m0;
      v16bf vb0a = frag_b_ld(vt0, NTOK);
      v16bf vb0b = frag_b_ld(vt0 + 32, NTOK);
      acc0 = wmma_bf16(pa0, vb0a, acc0);
      acc0 = wmma_bf16(pa1, vb0b, acc0);
      v16bf vb1a = frag_b_ld(vt1, NTOK);
      v16bf vb1b = frag_b_ld(vt1 + 32, NTOK);
      acc1 = wmma_bf16(pa0, vb1a, acc1);
      acc1 = wmma_bf16(pa1, vb1b, acc1);
    }
    __syncthreads();
  }

  const int b = bh >> 4, h = bh & 15;
#pragma unroll
  for (int e = 0; e < 2; ++e) {
    v8f acc = e ? acc1 : acc0;
    const int d = (di0 + e) * 16 + col;
#pragma unroll
    for (int r = 0; r < 8; ++r) {
      const int nl = ni * 16 + r + (hi << 3);
      const float z = fmaxf(Zrow[nl], 1e-12f);
      const float v = acc[r] / z;
      const int tok = b * NTOK + n0 + nl;
      ob[(size_t)tok * EMB + h * HD + d] = (bf16_t)v;
    }
  }
}

// ---------------------------------------------------------------------------
// Host-side orchestration
// ---------------------------------------------------------------------------
extern "C" void kernel_launch(void* const* d_in, const int* in_sizes, int n_in,
                              void* d_out, int out_size, void* d_ws, size_t ws_size,
                              hipStream_t stream) {
  const float* x  = (const float*)d_in[0];
  const float* wq = (const float*)d_in[1];
  const float* bq = (const float*)d_in[2];
  const float* wk = (const float*)d_in[3];
  const float* bk = (const float*)d_in[4];
  const float* wv = (const float*)d_in[5];
  const float* bv = (const float*)d_in[6];
  const float* wo = (const float*)d_in[7];
  const float* bo = (const float*)d_in[8];

  char* p = (char*)d_ws;
  auto carve = [&](size_t bytes) {
    void* r = (void*)p;
    p += (bytes + 255) & ~(size_t)255;
    return r;
  };
  const size_t actN = (size_t)MROWS * EMB;
  const size_t wN   = (size_t)EMB * EMB;
  bf16_t* xh  = (bf16_t*)carve(actN * 2);
  bf16_t* wqh = (bf16_t*)carve(wN * 2);
  bf16_t* wkh = (bf16_t*)carve(wN * 2);
  bf16_t* wvh = (bf16_t*)carve(wN * 2);
  bf16_t* woh = (bf16_t*)carve(wN * 2);
  bf16_t* qhd = (bf16_t*)carve(actN * 2);           // (B,H,N,D)
  bf16_t* khd = (bf16_t*)carve(actN * 2);           // (B,H,N,D)
  bf16_t* vTd = (bf16_t*)carve(actN * 2);           // (B,H,D,N)
  bf16_t* ob  = (bf16_t*)carve(actN * 2);           // (B*N, E)
  float*  q2  = (float*)carve((size_t)BH * NTOK * 4);
  float*  k2  = (float*)carve((size_t)BH * NTOK * 4);
  float*  ncol= (float*)carve((size_t)BH * NTOK * 4);

  cvt_bf16_kernel<<<(int)((actN + 255) / 256), 256, 0, stream>>>(x, xh, (int)actN);
  cvt_bf16_kernel<<<(int)((wN + 255) / 256), 256, 0, stream>>>(wq, wqh, (int)wN);
  cvt_bf16_kernel<<<(int)((wN + 255) / 256), 256, 0, stream>>>(wk, wkh, (int)wN);
  cvt_bf16_kernel<<<(int)((wN + 255) / 256), 256, 0, stream>>>(wv, wvh, (int)wN);
  cvt_bf16_kernel<<<(int)((wN + 255) / 256), 256, 0, stream>>>(wo, woh, (int)wN);

  dim3 ggrid(MROWS / 128, EMB / 64);
  gemm128_kernel<<<ggrid, 256, 0, stream>>>(xh, wqh, bq, qhd, 0);
  gemm128_kernel<<<ggrid, 256, 0, stream>>>(xh, wkh, bk, khd, 0);
  gemm128_kernel<<<ggrid, 256, 0, stream>>>(xh, wvh, bv, vTd, 1);

  const int nrows = BH * NTOK;
  rowsumsq_kernel<<<(nrows + 255) / 256, 256, 0, stream>>>(qhd, q2, nrows);
  rowsumsq_kernel<<<(nrows + 255) / 256, 256, 0, stream>>>(khd, k2, nrows);

  colsum_kernel<<<dim3(BH, NTOK / 64), 256, 0, stream>>>(qhd, khd, q2, k2, ncol);

  fusedpv_kernel<<<dim3(BH, NTOK / 64), 256, 0, stream>>>(qhd, khd, vTd, q2, k2,
                                                          ncol, ob);

  gemm128_kernel<<<ggrid, 256, 0, stream>>>(ob, woh, bo, d_out, 3);
}
